// MultiHeadAttention_16209206575251
// MI455X (gfx1250) — matrix operand order repaired and checked by reading, not compiled, not run on hardware
//
#include <hip/hip_runtime.h>
#include <hip/hip_bf16.h>
#include <math.h>

typedef __attribute__((ext_vector_type(16))) _Float16 v16h;
typedef __attribute__((ext_vector_type(8)))  _Float16 v8h;
typedef __attribute__((ext_vector_type(4)))  _Float16 v4h;
typedef __attribute__((ext_vector_type(8)))  float    v8f;

#define BATCH 2
#define NSEQ  2048
#define EMB   1024
#define NH    16
#define HD    64
#define MROWS (BATCH * NSEQ)   /* 4096 token rows */

// LDS row strides (halves): 16B-aligned for b128 ds ops, bank-conflict-free
#define KTS 72   /* K tile: 32 rows x 64 halves, stride 72 (144B = 36 dw) */
#define VTS 40   /* V tile: 64 rows x 32 halves, stride 40 (80B = 20 dw)  */
#define SBS 40   /* score shuffle: 16 rows x 32 halves, stride 40         */

// ---------------------------------------------------------------------------
// Fragment helpers (wave32; A/B k order is this file's contiguous one, not the ISA table; C per ISA)
// A 16x32 f16: lane row = lane%16; half h -> k=(lane/16)*16 + h
// B 32x16 f16: lane col = lane%16; half h -> k=(lane/16)*16 + h
// C 16x16 f32: lane col n = lane%16; vgpr r -> row m = r + 8*(lane/16)
// ---------------------------------------------------------------------------
__device__ __forceinline__ v16h combine16(v8h lo, v8h hi) {
  v16h a;
#pragma unroll
  for (int i = 0; i < 8; ++i) { a[i] = lo[i]; a[8 + i] = hi[i]; }
  return a;
}

// A from row-major f16 [*, lda]: rows m0..m0+15, cols k0..k0+31
__device__ __forceinline__ v16h load_a_frag(const _Float16* __restrict__ base,
                                            int lda, int m0, int k0, int lane) {
  int llo = lane & 15, lhi = lane >> 4;
  const _Float16* p = base + (size_t)(m0 + llo) * lda + k0 + lhi * 16;
  return *(const v16h*)p;
}

// B(k,n) = base[(n0+n)*ldb + k0+k]  (base holds N-major rows, contiguous k)
__device__ __forceinline__ v16h load_b_frag(const _Float16* __restrict__ base,
                                            int ldb, int n0, int k0, int lane) {
  int llo = lane & 15, lhi = lane >> 4;
  return *(const v16h*)(base + (size_t)(n0 + llo) * ldb + k0 + lhi * 16);
}

__device__ __forceinline__ v8f wmma_f16(v16h a, v16h b, v8f c) {
  return __builtin_amdgcn_wmma_f32_16x16x32_f16(false, a, false, b,
                                                (short)0, c, false, false);
}

// Branch-free exact-erf GELU (Abramowitz-Stegun 7.1.26, |err| < 1.5e-7):
// straight-line VALU, v_rcp_f32 + native v_exp_f32, no EXEC churn.
__device__ __forceinline__ float gelu_f(float x) {
  float z  = x * 0.70710678118f;                    // x / sqrt(2)
  float az = fabsf(z);
  float t  = __builtin_amdgcn_rcpf(fmaf(0.3275911f, az, 1.0f));
  float p  = fmaf(1.061405429f, t, -1.453152027f);
  p = fmaf(p, t, 1.421413741f);
  p = fmaf(p, t, -0.284496736f);
  p = fmaf(p, t, 0.254829592f);
  p = p * t;
  float e  = __expf(-az * az);                      // v_exp_f32
  float er = copysignf(fmaf(-p, e, 1.0f), z);
  return 0.5f * x * (1.0f + er);
}

// ---------------------------------------------------------------------------
// fp32 -> f16 conversion, 4 elements / thread (b128 load, b64 store)
// ---------------------------------------------------------------------------
__global__ void cvt_f32_f16_v4(const float4* __restrict__ src,
                               v4h* __restrict__ dst, int n4) {
  int i = blockIdx.x * blockDim.x + threadIdx.x;
  if (i < n4) {
    float4 f = src[i];
    v4h h;
    h[0] = (_Float16)f.x; h[1] = (_Float16)f.y;
    h[2] = (_Float16)f.z; h[3] = (_Float16)f.w;
    dst[i] = h;
  }
}

// ---------------------------------------------------------------------------
// GEMM: y[m,n] = sum_k A[m,k] * W[n,k] + bias[n]
// One wave computes a 16(M) x 64(N) strip: A fragment reused 4x per K-step.
// MODE 0: store f32 row-major; MODE 1: f16 row-major; MODE 2: f16 transposed.
// ---------------------------------------------------------------------------
template <int MODE>
__global__ void gemm_wmma(const _Float16* __restrict__ A,
                          const _Float16* __restrict__ W,
                          const float* __restrict__ bias,
                          void* __restrict__ Cout,
                          int M, int N, int K, int lda, int ldb, int ldc) {
  int lane = threadIdx.x & 31;
  int wave = threadIdx.x >> 5;
  int tile = blockIdx.x * 8 + wave;
  int ntn  = N >> 6;                  // 64-wide N strips
  int m0   = (tile / ntn) << 4;
  int n0   = (tile % ntn) << 6;
  if (m0 >= M) return;
  int llo = lane & 15, lhi = lane >> 4;

  v8f acc[4];
#pragma unroll
  for (int t = 0; t < 4; ++t) {
    float bv = bias[n0 + t * 16 + llo];
#pragma unroll
    for (int i = 0; i < 8; ++i) acc[t][i] = bv;   // bias broadcast along M
  }

  for (int k0 = 0; k0 < K; k0 += 32) {
    v16h a = load_a_frag(A, lda, m0, k0, lane);
    if (k0 + 32 < K)
      __builtin_prefetch(A + (size_t)(m0 + llo) * lda + k0 + 32, 0, 1);
#pragma unroll
    for (int t = 0; t < 4; ++t) {
      v16h b = load_b_frag(W, ldb, n0 + t * 16, k0, lane);
      acc[t] = wmma_f16(a, b, acc[t]);
    }
  }

#pragma unroll
  for (int t = 0; t < 4; ++t) {
    int nn = n0 + t * 16;
    if (MODE == 0) {
      float* out = (float*)Cout;
#pragma unroll
      for (int r = 0; r < 8; ++r)
        out[(size_t)(m0 + 8 * lhi + r) * ldc + nn + llo] = acc[t][r];
    } else if (MODE == 1) {
      _Float16* out = (_Float16*)Cout;
#pragma unroll
      for (int r = 0; r < 8; ++r)
        out[(size_t)(m0 + 8 * lhi + r) * ldc + nn + llo] = (_Float16)acc[t][r];
    } else {
      _Float16* out = (_Float16*)Cout;  // transposed: out[n, m]
      v8h h;
#pragma unroll
      for (int r = 0; r < 8; ++r) h[r] = (_Float16)acc[t][r];
      *(v8h*)(out + (size_t)(nn + llo) * ldc + m0 + 8 * lhi) = h;
    }
  }
}

// ---------------------------------------------------------------------------
// Attention: block = 8 waves, all in one (b,h); wave owns a 16-row i-tile.
// Per 32-wide j-step:
//   - cooperatively stage K tile (32x64) and V^T tile (64x32) in LDS
//     (1 coalesced b128 per thread per tile; was 8x redundant per-wave loads)
//   - scores = q @ k^T from LDS B-frags; gelu -> weights (HBM) + f16 shuffle
//   - ctx += w_f16 @ v from LDS B-frags (flash-style, no weights re-read)
// ---------------------------------------------------------------------------
__global__ void attn_wmma(const _Float16* __restrict__ qh,
                          const _Float16* __restrict__ kh,
                          const _Float16* __restrict__ vT,
                          float* __restrict__ wout,
                          _Float16* __restrict__ ctx) {
  __shared__ _Float16 kt[32 * KTS];        // ~4.5 KB
  __shared__ _Float16 vt[64 * VTS];        // 5 KB
  __shared__ _Float16 sbuf[8 * 16 * SBS];  // 10 KB, per-wave score shuffle

  int tid  = threadIdx.x;
  int lane = tid & 31;
  int wave = tid >> 5;
  int llo = lane & 15, lhi = lane >> 4;

  int tile = blockIdx.x * 8 + wave;   // 0..4095
  int it = tile & 127;                // i-tile index within (b,h)
  int bh = tile >> 7;                 // 0..31 (shared by all waves of a block)
  int b = bh >> 4, h = bh & 15;
  int i0 = it << 4;
  int rowbase = b * NSEQ;
  _Float16* my = sbuf + wave * (16 * SBS);

  // cooperative-stage addressing (per thread, constant over the j-loop)
  int kr = tid >> 3, ko = (tid & 7) * 8;   // K tile: 32 rows x 8 chunks
  int vr = tid >> 2, vo = (tid & 3) * 8;   // V tile: 64 rows x 4 chunks
  const _Float16* kgp = kh + (size_t)(rowbase + kr) * EMB + h * HD + ko;
  const _Float16* vgp = vT + (size_t)(h * HD + vr) * MROWS + rowbase + vo;
  _Float16* klp = kt + kr * KTS + ko;
  _Float16* vlp = vt + vr * VTS + vo;

  // Q fragments held in registers for all 128 j-tiles
  v16h qa0 = load_a_frag(qh, EMB, rowbase + i0, h * HD + 0,  lane);
  v16h qa1 = load_a_frag(qh, EMB, rowbase + i0, h * HD + 32, lane);

  v8f cacc[4];
#pragma unroll
  for (int t = 0; t < 4; ++t)
#pragma unroll
    for (int r = 0; r < 8; ++r) cacc[t][r] = 0.f;

  const float scale = 0.125f;                 // 1/sqrt(HD)
  size_t wbase = (size_t)bh * NSEQ * NSEQ;

  for (int j0 = 0; j0 < NSEQ; j0 += 32) {
    // stage K/V tiles for this j-step (coalesced global -> LDS)
    *(v8h*)klp = *(const v8h*)(kgp + (size_t)j0 * EMB);
    *(v8h*)vlp = *(const v8h*)(vgp + j0);
    __syncthreads();

#pragma unroll
    for (int half = 0; half < 2; ++half) {
      int jj = j0 + half * 16;
      v8f s;
#pragma unroll
      for (int r = 0; r < 8; ++r) s[r] = 0.f;
      v16h kb0 = *(const v16h*)(kt + (half * 16 + llo) * KTS + lhi * 16);
      s = wmma_f16(qa0, kb0, s);
      v16h kb1 = *(const v16h*)(kt + (half * 16 + llo) * KTS + 32 + lhi * 16);
      s = wmma_f16(qa1, kb1, s);
#pragma unroll
      for (int r = 0; r < 8; ++r) {
        float g = gelu_f(s[r] * scale);
        wout[wbase + (size_t)(i0 + 8 * lhi + r) * NSEQ + jj + llo] = g;
        my[(8 * lhi + r) * SBS + half * 16 + llo] = (_Float16)g;
      }
    }

    // A-layout fragment of the gelu'd scores (same-wave DS, in-order pipe)
    v8h wl0 = *(const v8h*)(my + llo * SBS + lhi * 16);
    v8h wl1 = *(const v8h*)(my + llo * SBS + lhi * 16 + 8);
    v16h wa = combine16(wl0, wl1);

#pragma unroll
    for (int t = 0; t < 4; ++t) {
      v16h vb = *(const v16h*)(vt + (t * 16 + llo) * VTS + lhi * 16);
      cacc[t] = wmma_f16(wa, vb, cacc[t]);
    }
    __syncthreads();   // all reads of kt/vt done before next stage
  }

#pragma unroll
  for (int t = 0; t < 4; ++t)
#pragma unroll
    for (int r = 0; r < 8; ++r)
      ctx[(size_t)(rowbase + i0 + 8 * lhi + r) * EMB + h * HD + t * 16 + llo] =
          (_Float16)cacc[t][r];
}

// ---------------------------------------------------------------------------
// Launch
// inputs: x, Wq, bq, Wk, bk, Wv, bv, Wo, bo
// d_out:  out [2,2048,1024] f32  ++  weights [2,16,2048,2048] f32
// ---------------------------------------------------------------------------
extern "C" void kernel_launch(void* const* d_in, const int* in_sizes, int n_in,
                              void* d_out, int out_size, void* d_ws, size_t ws_size,
                              hipStream_t stream) {
  const float* x  = (const float*)d_in[0];
  const float* Wq = (const float*)d_in[1];
  const float* bq = (const float*)d_in[2];
  const float* Wk = (const float*)d_in[3];
  const float* bk = (const float*)d_in[4];
  const float* Wv = (const float*)d_in[5];
  const float* bv = (const float*)d_in[6];
  const float* Wo = (const float*)d_in[7];
  const float* bo = (const float*)d_in[8];

  char* ws = (char*)d_ws;
  const size_t MB = 1ull << 20;
  _Float16* xh  = (_Float16*)(ws + 0 * MB);   // 8 MiB, reused as ctx later
  _Float16* qh  = (_Float16*)(ws + 8 * MB);   // 8 MiB
  _Float16* kh  = (_Float16*)(ws + 16 * MB);  // 8 MiB
  _Float16* vth = (_Float16*)(ws + 24 * MB);  // 8 MiB  (transposed [EMB, MROWS])
  _Float16* wqh = (_Float16*)(ws + 32 * MB);  // 2 MiB each
  _Float16* wkh = (_Float16*)(ws + 34 * MB);
  _Float16* wvh = (_Float16*)(ws + 36 * MB);
  _Float16* woh = (_Float16*)(ws + 38 * MB);  // total 40 MiB
  _Float16* ch  = xh;                         // ctx overwrites xh (dead after QKV)

  float* outp = (float*)d_out;                       // [4096, 1024]
  float* wts  = outp + (size_t)MROWS * EMB;          // [2,16,2048,2048]

  dim3 blk(256);
  cvt_f32_f16_v4<<<(MROWS * EMB / 4) / 256, blk, 0, stream>>>(
      (const float4*)x, (v4h*)xh, MROWS * EMB / 4);
  cvt_f32_f16_v4<<<(EMB * EMB / 4) / 256, blk, 0, stream>>>(
      (const float4*)Wq, (v4h*)wqh, EMB * EMB / 4);
  cvt_f32_f16_v4<<<(EMB * EMB / 4) / 256, blk, 0, stream>>>(
      (const float4*)Wk, (v4h*)wkh, EMB * EMB / 4);
  cvt_f32_f16_v4<<<(EMB * EMB / 4) / 256, blk, 0, stream>>>(
      (const float4*)Wv, (v4h*)wvh, EMB * EMB / 4);
  cvt_f32_f16_v4<<<(EMB * EMB / 4) / 256, blk, 0, stream>>>(
      (const float4*)Wo, (v4h*)woh, EMB * EMB / 4);

  int gtiles = (MROWS / 16) * (EMB / 64) / 8;   // 512 blocks of 8 waves
  gemm_wmma<1><<<gtiles, blk, 0, stream>>>(xh, wqh, bq, qh,  MROWS, EMB, EMB, EMB, EMB, EMB);
  gemm_wmma<1><<<gtiles, blk, 0, stream>>>(xh, wkh, bk, kh,  MROWS, EMB, EMB, EMB, EMB, EMB);
  gemm_wmma<2><<<gtiles, blk, 0, stream>>>(xh, wvh, bv, vth, MROWS, EMB, EMB, EMB, EMB, MROWS);

  attn_wmma<<<(BATCH * NH * (NSEQ / 16)) / 8, blk, 0, stream>>>(qh, kh, vth, wts, ch);

  gemm_wmma<0><<<gtiles, blk, 0, stream>>>(ch, woh, bo, outp, MROWS, EMB, EMB, EMB, EMB, EMB);
}
